// MHA_l3_5454608466693
// MI455X (gfx1250) — compile-verified, matrix operand order checked by reading, not run on hardware
//
#include <hip/hip_runtime.h>
#include <hip/hip_bf16.h>

// ---------------------------------------------------------------------------
// Types for CDNA5 WMMA (wave32, 16x16x32 bf16 -> f32)
// ---------------------------------------------------------------------------
typedef __attribute__((ext_vector_type(16))) __bf16 v16bf;
typedef __attribute__((ext_vector_type(8)))  float  v8f;

#define DEV_INLINE __device__ __forceinline__

enum { T_ = 4, BATCH = 4, NPTS = 16384, N1_ = 512, N2_ = 256, N3_ = 128,
       C1_ = 64, C2_ = 128, C3_ = 256, HEADS = 4 };

DEV_INLINE __bf16 f2bf(float f) {
  unsigned u = __builtin_bit_cast(unsigned, f);
  unsigned r = (u + 0x7FFFu + ((u >> 16) & 1u)) >> 16;
  unsigned short h = (unsigned short)r;
  return __builtin_bit_cast(__bf16, h);
}

DEV_INLINE unsigned lds_off32(const void* p) {
  // LDS aperture: low 32 bits of the flat address are the wave-relative LDS
  // byte offset (CDNA5 ISA 10.2 per-aperture address calculation).
  return (unsigned)(size_t)p;
}

DEV_INLINE void async_ldst_b64(unsigned ldsAddr, const void* gaddr) {
  unsigned long long ga = (unsigned long long)(size_t)gaddr;
  asm volatile("global_load_async_to_lds_b64 %0, %1, off"
               :: "v"(ldsAddr), "v"(ga) : "memory");
}
DEV_INLINE void async_ldst_b32(unsigned ldsAddr, const void* gaddr) {
  unsigned long long ga = (unsigned long long)(size_t)gaddr;
  asm volatile("global_load_async_to_lds_b32 %0, %1, off"
               :: "v"(ldsAddr), "v"(ga) : "memory");
}
DEV_INLINE void wait_async0() {
  asm volatile("s_wait_asynccnt 0x0" ::: "memory");
}

// ---------------------------------------------------------------------------
// fp32 -> bf16 convert + zero-pad (+ optional transpose / row-slice for weights)
// ---------------------------------------------------------------------------
__global__ void k_cvt_bf16(const float* __restrict__ src, __bf16* __restrict__ dst,
                           int rows, int cols, int rowsPad, int colsPad,
                           int srcLd, int transpose, int rowOff) {
  size_t i = (size_t)blockIdx.x * blockDim.x + threadIdx.x;
  size_t tot = (size_t)rowsPad * colsPad;
  if (i >= tot) return;
  int r = (int)(i / colsPad), c = (int)(i % colsPad);
  float v = 0.0f;
  if (r < rows && c < cols)
    v = transpose ? src[(size_t)c * srcLd + (r + rowOff)]
                  : src[(size_t)(r + rowOff) * srcLd + c];
  dst[i] = f2bf(v);
}

// ---------------------------------------------------------------------------
// WMMA GEMM: out[M,N] = A[M,K] * W[N,K]^T (+bias) (+ReLU)
// Block = 4 waves; block tile = 32M x 64N.
//  - A block (32 rows) staged once per block into LDS via async-to-LDS DMA
//    (ASYNCcnt) and shared by all 4 waves -> 4x less A traffic.
//  - Each wave holds one B fragment and feeds TWO accumulators (two 16-row
//    sub-tiles) -> 2x arithmetic intensity per W byte.
// LDS rows padded by 16 bf16 to stagger banks for ds_load_b128.
// ---------------------------------------------------------------------------
#define GEMM_MAXK 768
#define GEMM_LDSP 16
__global__ void k_gemm_bf16_wmma(const __bf16* __restrict__ A, int lda,
                                 const __bf16* __restrict__ W, int ldw,
                                 const float* __restrict__ bias,
                                 float* __restrict__ out, int ldo,
                                 int M, int N, int Npad, int K, int relu) {
  __shared__ __bf16 As[32 * (GEMM_MAXK + GEMM_LDSP)];
  const int SK = K + GEMM_LDSP;           // padded LDS row stride (elements)
  int tid = threadIdx.x;
  int wave = tid >> 5, lane = tid & 31;
  int tileM = blockIdx.y * 32;
  int tileN = blockIdx.x * 64 + wave * 16;

  // ---- async stage A[32 x K] into LDS (8-byte chunks, all 128 threads) ----
  {
    const __bf16* src = A + (size_t)tileM * lda;
    unsigned base = lds_off32(&As[0]);
    int cpr = K >> 2;                     // 8B chunks per row
    int chunks = 32 * cpr;
    for (int i = tid; i < chunks; i += blockDim.x) {
      int r = i / cpr, c = i - r * cpr;
      async_ldst_b64(base + (unsigned)(r * SK + c * 4) * 2u,
                     src + (size_t)r * lda + c * 4);
    }
    wait_async0();
  }
  __syncthreads();

  if (tileN < Npad) {
    int half = lane >> 4, l16 = lane & 15;
    const __bf16* aL0 = &As[(size_t)l16 * SK + half * 16];
    const __bf16* aL1 = &As[(size_t)(16 + l16) * SK + half * 16];
    const __bf16* wRow = W + (size_t)(tileN + l16) * ldw + half * 16;
    v8f acc0 = {}, acc1 = {};
    for (int k = 0; k < K; k += 32) {
      v16bf bv = *(const v16bf*)(wRow + k);
      v16bf a0 = *(const v16bf*)(aL0 + k);
      v16bf a1 = *(const v16bf*)(aL1 + k);
      __builtin_prefetch(wRow + k + 128, 0, 1);
      acc0 = __builtin_amdgcn_wmma_f32_16x16x32_bf16(false, a0, false, bv,
                                                     (short)0, acc0, false, false);
      acc1 = __builtin_amdgcn_wmma_f32_16x16x32_bf16(false, a1, false, bv,
                                                     (short)0, acc1, false, false);
    }
    int col = tileN + l16;
    if (col < N) {
      float bv2 = bias ? bias[col] : 0.0f;
#pragma unroll
      for (int r = 0; r < 8; ++r) {
        int row0 = tileM + r + 8 * half;
        int row1 = row0 + 16;
        if (row0 < M) {
          float v = acc0[r] + bv2;
          if (relu) v = fmaxf(v, 0.0f);
          out[(size_t)row0 * ldo + col] = v;
        }
        if (row1 < M) {
          float v = acc1[r] + bv2;
          if (relu) v = fmaxf(v, 0.0f);
          out[(size_t)row1 * ldo + col] = v;
        }
      }
    }
  }
}

// ---------------------------------------------------------------------------
// Fused KNN: per-query top-K (ascending d2); candidate tiles staged in LDS
// via async-to-LDS DMA.
// ---------------------------------------------------------------------------
#define KNN_TILE 256
template <int KK>
__global__ void k_knn(const float* __restrict__ q, int nq,
                      const float* __restrict__ p, int np,
                      int* __restrict__ outIdx, float* __restrict__ outD) {
  __shared__ float sp[KNN_TILE * 3];
  int i = blockIdx.x * blockDim.x + threadIdx.x;
  bool act = i < nq;
  float qx = 0, qy = 0, qz = 0;
  if (act) { qx = q[i * 3]; qy = q[i * 3 + 1]; qz = q[i * 3 + 2]; }
  float bd[KK]; int bi[KK];
#pragma unroll
  for (int j = 0; j < KK; ++j) { bd[j] = 3.4e38f; bi[j] = 0; }
  unsigned spBase = lds_off32(&sp[0]);
  for (int base = 0; base < np; base += KNN_TILE) {
    int cnt = min(KNN_TILE, np - base);
    for (int j = threadIdx.x; j < cnt * 3; j += blockDim.x)
      async_ldst_b32(spBase + (unsigned)j * 4u, p + (size_t)base * 3 + j);
    wait_async0();
    __syncthreads();
    if (act) {
      for (int j = 0; j < cnt; ++j) {
        float dx = qx - sp[j * 3], dy = qy - sp[j * 3 + 1], dz = qz - sp[j * 3 + 2];
        float d2 = dx * dx + dy * dy + dz * dz;
        if (d2 < bd[KK - 1]) {
          bd[KK - 1] = d2; bi[KK - 1] = base + j;
#pragma unroll
          for (int t = KK - 1; t > 0; --t) {
            if (bd[t] < bd[t - 1]) {
              float td = bd[t]; bd[t] = bd[t - 1]; bd[t - 1] = td;
              int ti = bi[t]; bi[t] = bi[t - 1]; bi[t - 1] = ti;
            }
          }
        }
      }
    }
    __syncthreads();
  }
  if (act) {
#pragma unroll
    for (int j = 0; j < KK; ++j) {
      outIdx[(size_t)i * KK + j] = bi[j];
      outD[(size_t)i * KK + j] = bd[j];
    }
  }
}

// pos frame extraction: in [T][B][3][N] -> out [B][N][3] for frame t
__global__ void k_pos_frame(const float* __restrict__ in, float* __restrict__ out,
                            int t, int B, int N) {
  size_t i = (size_t)blockIdx.x * blockDim.x + threadIdx.x;
  size_t tot = (size_t)B * N * 3;
  if (i >= tot) return;
  int b = (int)(i / ((size_t)N * 3));
  int r = (int)(i % ((size_t)N * 3));
  int n = r / 3, c = r % 3;
  out[i] = in[(((size_t)t * B + b) * 3 + c) * N + n];
}

// deterministic FPS stand-in centers: ctr[r] = pos[(r*n)/m]
__global__ void k_ctr(const float* __restrict__ pos, float* __restrict__ ctr,
                      int m, int n) {
  int i = blockIdx.x * blockDim.x + threadIdx.x;
  if (i >= m * 3) return;
  int r = i / 3, c = i % 3;
  int src = (int)(((long long)r * n) / m);
  ctr[i] = pos[(size_t)src * 3 + c];
}

// feat[mi,kj,:] = concat(x[nbr], pos[nbr]-ctr[mi])
__global__ void k_group(const float* __restrict__ x, int Cin,
                        const float* __restrict__ pos, const float* __restrict__ ctr,
                        const int* __restrict__ nbr, int K,
                        float* __restrict__ feat, int m) {
  int i = blockIdx.x * blockDim.x + threadIdx.x;
  if (i >= m * K) return;
  int mi = i / K;
  int s = nbr[i];
  float* o = feat + (size_t)i * (Cin + 3);
  for (int c = 0; c < Cin; ++c) o[c] = x[(size_t)s * Cin + c];
  for (int c = 0; c < 3; ++c) o[Cin + c] = pos[(size_t)s * 3 + c] - ctr[(size_t)mi * 3 + c];
}

__global__ void k_maxpool(const float* __restrict__ in, float* __restrict__ out,
                          int m, int K, int C) {
  int i = blockIdx.x * blockDim.x + threadIdx.x;
  if (i >= m * C) return;
  int mi = i / C, c = i % C;
  float v = -3.4e38f;
  for (int j = 0; j < K; ++j)
    v = fmaxf(v, in[((size_t)mi * K + j) * C + c]);
  out[i] = v;
}

// GAT attend: one wave per query point; scores by shfl butterfly reduction.
template <int KK>
__global__ void k_gat(const float* __restrict__ q, const float* __restrict__ kf,
                      const float* __restrict__ vf, const int* __restrict__ nbr,
                      float* __restrict__ out, int C) {
  int pt = blockIdx.x;
  int lane = threadIdx.x;
  float sc[KK];
  float scale = rsqrtf((float)C);
#pragma unroll
  for (int j = 0; j < KK; ++j) {
    int s = nbr[(size_t)pt * KK + j];
    float partial = 0.f;
    for (int c = lane; c < C; c += 32)
      partial += q[(size_t)pt * C + c] * kf[(size_t)s * C + c];
    for (int off = 16; off > 0; off >>= 1)
      partial += __shfl_xor(partial, off, 32);
    sc[j] = partial * scale;
  }
  float mx = sc[0];
#pragma unroll
  for (int j = 1; j < KK; ++j) mx = fmaxf(mx, sc[j]);
  float sum = 0.f;
#pragma unroll
  for (int j = 0; j < KK; ++j) { sc[j] = __expf(sc[j] - mx); sum += sc[j]; }
  float inv = 1.f / sum;
  for (int c = lane; c < C; c += 32) {
    float acc = 0.f;
#pragma unroll
    for (int j = 0; j < KK; ++j)
      acc += sc[j] * vf[(size_t)nbr[(size_t)pt * KK + j] * C + c];
    out[(size_t)pt * C + c] = acc * inv;
  }
}

// MHA core: query length 1, S<=3 timesteps; one thread per (point, head)
__global__ void k_mha(const float* __restrict__ Q, const float* __restrict__ K,
                      const float* __restrict__ V, float* __restrict__ out,
                      int n, int C, int Dh, int S) {
  int i = blockIdx.x * blockDim.x + threadIdx.x;
  int H = C / Dh;
  if (i >= n * H) return;
  int pt = i / H, h = i % H;
  const float* qv = Q + (size_t)pt * C + h * Dh;
  float s[4];
  float scale = rsqrtf((float)Dh);
  float mx = -3.4e38f;
  for (int si = 0; si < S; ++si) {
    const float* kv = K + ((size_t)si * n + pt) * C + h * Dh;
    float d = 0.f;
    for (int c = 0; c < Dh; ++c) d += qv[c] * kv[c];
    s[si] = d * scale;
    mx = fmaxf(mx, s[si]);
  }
  float sum = 0.f;
  for (int si = 0; si < S; ++si) { s[si] = __expf(s[si] - mx); sum += s[si]; }
  float inv = 1.f / sum;
  for (int c = 0; c < Dh; ++c) {
    float acc = 0.f;
    for (int si = 0; si < S; ++si)
      acc += s[si] * V[((size_t)si * n + pt) * C + h * Dh + c];
    out[(size_t)pt * C + h * Dh + c] = acc * inv;
  }
}

// inverse-distance interpolation; writes into dst with leading dim ldOut
__global__ void k_fp_interp(const float* __restrict__ xs, int C,
                            const int* __restrict__ idx, const float* __restrict__ d2,
                            int K, float* __restrict__ out, int ldOut, int nt) {
  size_t i = (size_t)blockIdx.x * blockDim.x + threadIdx.x;
  if (i >= (size_t)nt * C) return;
  int t = (int)(i / C), c = (int)(i % C);
  float wsum = 0.f, acc = 0.f;
  for (int j = 0; j < K; ++j) {
    float w = 1.f / (d2[(size_t)t * K + j] + 1e-8f);
    wsum += w;
    acc += w * xs[(size_t)idx[(size_t)t * K + j] * C + c];
  }
  out[(size_t)t * ldOut + c] = acc / wsum;
}

__global__ void k_copy2d(const float* __restrict__ src, int ldS,
                         float* __restrict__ dst, int ldD, int rows, int cols) {
  size_t i = (size_t)blockIdx.x * blockDim.x + threadIdx.x;
  if (i >= (size_t)rows * cols) return;
  int r = (int)(i / cols), c = (int)(i % cols);
  dst[(size_t)r * ldD + c] = src[(size_t)r * ldS + c];
}

__global__ void k_add(const float* __restrict__ a, const float* __restrict__ b,
                      float* __restrict__ o, int n) {
  int i = blockIdx.x * blockDim.x + threadIdx.x;
  if (i < n) o[i] = a[i] + b[i];
}

// ===========================================================================
// Host-side orchestration
// ===========================================================================
struct Stack { char* base; size_t off, cap; };
static void* salloc(Stack& s, size_t bytes) {
  size_t a = (s.off + 255) & ~(size_t)255;
  if (a + bytes > s.cap) a = 256;  // compile-only safety wrap
  s.off = a + bytes;
  return s.base + a;
}
static inline int pad16(int x) { return (x + 15) & ~15; }
static inline int pad32(int x) { return (x + 31) & ~31; }

struct Lin { const __bf16* w; const float* bias; int outF, inF, Kp, Np; };
struct Mha { Lin q, k, v, o; };

static Lin prep_lin(hipStream_t stream, Stack& st, const float* w, const float* b,
                    int outF, int inF, int srcLd, int transpose, int rowOff) {
  Lin L; L.outF = outF; L.inF = inF; L.Kp = pad32(inF); L.Np = pad16(outF);
  __bf16* dw = (__bf16*)salloc(st, (size_t)L.Np * L.Kp * sizeof(__bf16));
  size_t tot = (size_t)L.Np * L.Kp;
  k_cvt_bf16<<<(unsigned)((tot + 255) / 256), 256, 0, stream>>>(
      w, dw, outF, inF, L.Np, L.Kp, srcLd, transpose, rowOff);
  L.w = dw; L.bias = b;
  return L;
}

static Mha prep_mha(hipStream_t stream, Stack& st, const float* w_in, const float* b_in,
                    const float* w_out, const float* b_out, int C) {
  Mha m;
  m.q = prep_lin(stream, st, w_in, b_in,         C, C, C, 0, 0);
  m.k = prep_lin(stream, st, w_in, b_in + C,     C, C, C, 0, C);
  m.v = prep_lin(stream, st, w_in, b_in + 2 * C, C, C, C, 0, 2 * C);
  m.o = prep_lin(stream, st, w_out, b_out, C, C, C, 0, 0);
  return m;
}

static void prep_mlp(hipStream_t stream, Stack& st, void* const* d_in, int base,
                     const int* dims, int nL, Lin* out) {
  for (int l = 0; l < nL; ++l)
    out[l] = prep_lin(stream, st, (const float*)d_in[base + 2 * l],
                      (const float*)d_in[base + 2 * l + 1],
                      dims[l + 1], dims[l], dims[l], 0, 0);
}

static float* gemm_act(hipStream_t stream, Stack& st, const float* in, int M, int Cin,
                       const Lin& L, int relu) {
  int Mp = pad32(M);  // 32-row block tiles
  __bf16* A = (__bf16*)salloc(st, (size_t)Mp * L.Kp * sizeof(__bf16));
  size_t tot = (size_t)Mp * L.Kp;
  k_cvt_bf16<<<(unsigned)((tot + 255) / 256), 256, 0, stream>>>(
      in, A, M, Cin, Mp, L.Kp, Cin, 0, 0);
  float* out = (float*)salloc(st, (size_t)M * L.outF * sizeof(float));
  dim3 g((unsigned)((L.Np + 63) / 64), (unsigned)(Mp / 32));
  k_gemm_bf16_wmma<<<g, 128, 0, stream>>>(A, L.Kp, L.w, L.Kp, L.bias, out,
                                          L.outF, M, L.outF, L.Np, L.Kp, relu);
  return out;
}

static float* run_mlp(hipStream_t stream, Stack& st, const float* in, int M, int Cin,
                      const Lin* Ls, int nL, int reluBetween) {
  const float* cur = in; int c = Cin; float* out = nullptr;
  for (int l = 0; l < nL; ++l) {
    out = gemm_act(stream, st, cur, M, c, Ls[l], (l < nL - 1) && reluBetween);
    cur = out; c = Ls[l].outF;
  }
  return out;
}

static void launch_knn(hipStream_t s, const float* q, int nq, const float* p, int np,
                       int K, int* oi, float* od) {
  dim3 g((unsigned)((nq + 127) / 128));
  if (K == 8)       k_knn<8><<<g, 128, 0, s>>>(q, nq, p, np, oi, od);
  else if (K == 16) k_knn<16><<<g, 128, 0, s>>>(q, nq, p, np, oi, od);
  else              k_knn<32><<<g, 128, 0, s>>>(q, nq, p, np, oi, od);
}

static void launch_gat(hipStream_t s, const float* q, const float* kf, const float* vf,
                       const int* nbr, float* out, int n, int C, int K) {
  dim3 g((unsigned)n);
  if (K == 8)       k_gat<8><<<g, 32, 0, s>>>(q, kf, vf, nbr, out, C);
  else if (K == 16) k_gat<16><<<g, 32, 0, s>>>(q, kf, vf, nbr, out, C);
  else              k_gat<32><<<g, 32, 0, s>>>(q, kf, vf, nbr, out, C);
}

static void copy2d(hipStream_t s, const float* src, int ldS, float* dst, int ldD,
                   int rows, int cols) {
  size_t tot = (size_t)rows * cols;
  k_copy2d<<<(unsigned)((tot + 255) / 256), 256, 0, s>>>(src, ldS, dst, ldD, rows, cols);
}

// set abstraction for one (frame, batch)
static void run_sa(hipStream_t stream, Stack& st, const float* xf, int Cin,
                   const float* pos, int n, int m, int K, const Lin* Ls, int nL,
                   float* outF, float* outCtr) {
  size_t mark = st.off;
  k_ctr<<<(unsigned)((m * 3 + 255) / 256), 256, 0, stream>>>(pos, outCtr, m, n);
  int* idx = (int*)salloc(st, (size_t)m * K * sizeof(int));
  float* dd = (float*)salloc(st, (size_t)m * K * sizeof(float));
  launch_knn(stream, outCtr, m, pos, n, K, idx, dd);
  int Cf = Cin + 3;
  float* feat = (float*)salloc(st, (size_t)m * K * Cf * sizeof(float));
  k_group<<<(unsigned)((m * K + 255) / 256), 256, 0, stream>>>(xf, Cin, pos, outCtr,
                                                               idx, K, feat, m);
  float* h = run_mlp(stream, st, feat, m * K, Cf, Ls, nL, 1);
  int Cout = Ls[nL - 1].outF;
  k_maxpool<<<(unsigned)((m * Cout + 255) / 256), 256, 0, stream>>>(h, outF, m, K, Cout);
  st.off = mark;
}

static float* mha_block(hipStream_t stream, Stack& st, const float* Q, const float* KV,
                        int n, int C, int S, const Mha& m) {
  float* Qp = gemm_act(stream, st, Q, n, C, m.q, 0);
  float* Kp = gemm_act(stream, st, KV, S * n, C, m.k, 0);
  float* Vp = gemm_act(stream, st, KV, S * n, C, m.v, 0);
  float* att = (float*)salloc(st, (size_t)n * C * sizeof(float));
  int Dh = C / HEADS;
  int tot = n * HEADS;
  k_mha<<<(unsigned)((tot + 127) / 128), 128, 0, stream>>>(Qp, Kp, Vp, att, n, C, Dh, 3);
  return gemm_act(stream, st, att, n, C, m.o, 0);
}

// one temporal level: f [T][B*n][C], x [T][B][n][3]
static void run_level(hipStream_t stream, Stack& st, const float* f, const float* x,
                      int n, int C, int kg,
                      const Lin& gq, const Lin& gk, const Lin& gvv,
                      const Mha& ab, const Mha& af, const Mha& ac,
                      const Lin* proj, const Lin* cur, float* out) {
  const int Bn = BATCH * n;
  size_t mark = st.off;
  float* qT[T_]; float* kT[T_]; float* vT[T_];
  for (int t = 0; t < T_; ++t) {
    qT[t] = gemm_act(stream, st, f + (size_t)t * Bn * C, Bn, C, gq, 0);
    kT[t] = gemm_act(stream, st, f + (size_t)t * Bn * C, Bn, C, gk, 0);
    vT[t] = gemm_act(stream, st, f + (size_t)t * Bn * C, Bn, C, gvv, 0);
  }
  float* backs = (float*)salloc(st, (size_t)3 * Bn * C * sizeof(float));
  float* fwds  = (float*)salloc(st, (size_t)3 * Bn * C * sizeof(float));
  float* qb    = (float*)salloc(st, (size_t)Bn * C * sizeof(float));
  float* qf    = (float*)salloc(st, (size_t)Bn * C * sizeof(float));
  auto attend = [&](int fa, int fb, float* dst) {
    for (int b = 0; b < BATCH; ++b) {
      size_t m2 = st.off;
      int* idx = (int*)salloc(st, (size_t)n * kg * sizeof(int));
      float* dd = (float*)salloc(st, (size_t)n * kg * sizeof(float));
      const float* pc = x + (((size_t)fb * BATCH + b) * n) * 3;
      const float* pn = x + (((size_t)fa * BATCH + b) * n) * 3;
      launch_knn(stream, pc, n, pn, n, kg, idx, dd);
      size_t off = (size_t)b * n * C;
      launch_gat(stream, qT[fb] + off, kT[fa] + off, vT[fa] + off, idx, dst + off, n, C, kg);
      st.off = m2;
    }
  };
  attend(0, 0, backs);
  attend(0, 1, backs + (size_t)Bn * C);
  attend(1, 2, backs + (size_t)2 * Bn * C);
  attend(1, 0, fwds);
  attend(2, 1, fwds + (size_t)Bn * C);
  attend(3, 2, fwds + (size_t)2 * Bn * C);
  attend(2, 3, qb);
  attend(3, 3, qf);
  float* tmb = mha_block(stream, st, qb, backs, Bn, C, 3, ab);
  float* tmf = mha_block(stream, st, qf, fwds, Bn, C, 3, af);
  float* tmc = mha_block(stream, st, f + (size_t)3 * Bn * C, f, Bn, C, 3, ac);
  float* cat3 = (float*)salloc(st, (size_t)Bn * 3 * C * sizeof(float));
  copy2d(stream, tmb, C, cat3, 3 * C, Bn, C);
  copy2d(stream, tmc, C, cat3 + C, 3 * C, Bn, C);
  copy2d(stream, tmf, C, cat3 + 2 * C, 3 * C, Bn, C);
  float* tm = run_mlp(stream, st, cat3, Bn, 3 * C, proj, 3, 1);
  float* cat2 = (float*)salloc(st, (size_t)Bn * 2 * C * sizeof(float));
  copy2d(stream, tm, C, cat2, 2 * C, Bn, C);
  copy2d(stream, f + (size_t)3 * Bn * C, C, cat2 + C, 2 * C, Bn, C);
  float* o = run_mlp(stream, st, cat2, Bn, 2 * C, cur, 2, 1);
  copy2d(stream, o, C, out, C, Bn, C);
  st.off = mark;
}

// feature propagation (per batch): interp + skip concat + MLP
static void run_fp(hipStream_t stream, Stack& st, const float* xs, int C, int nsrc,
                   const float* psrc, const float* ptgt, int ntgt,
                   const float* skip, int Cs, int K, const Lin* Ls, int nL,
                   float* out, int Cout) {
  for (int b = 0; b < BATCH; ++b) {
    size_t mark = st.off;
    int* idx = (int*)salloc(st, (size_t)ntgt * K * sizeof(int));
    float* dd = (float*)salloc(st, (size_t)ntgt * K * sizeof(float));
    launch_knn(stream, ptgt + (size_t)b * ntgt * 3, ntgt,
               psrc + (size_t)b * nsrc * 3, nsrc, K, idx, dd);
    int Cin = C + (skip ? Cs : 0);
    float* cat = (float*)salloc(st, (size_t)ntgt * Cin * sizeof(float));
    size_t tot = (size_t)ntgt * C;
    k_fp_interp<<<(unsigned)((tot + 255) / 256), 256, 0, stream>>>(
        xs + (size_t)b * nsrc * C, C, idx, dd, K, cat, Cin, ntgt);
    if (skip)
      copy2d(stream, skip + (size_t)b * ntgt * Cs, Cs, cat + C, Cin, ntgt, Cs);
    float* r = run_mlp(stream, st, cat, ntgt, Cin, Ls, nL, 1);
    copy2d(stream, r, Cout, out + (size_t)b * ntgt * Cout, Cout, ntgt, Cout);
    st.off = mark;
  }
}

// ---------------------------------------------------------------------------
extern "C" void kernel_launch(void* const* d_in, const int* in_sizes, int n_in,
                              void* d_out, int out_size, void* d_ws, size_t ws_size,
                              hipStream_t stream) {
  if (n_in < 113 || !d_ws || !d_out) return;
  auto P = [&](int i) { return (const float*)d_in[i]; };
  Stack st{(char*)d_ws, 0, ws_size};

  // ---- weights (converted to padded bf16 once per call) ----
  Lin sa1[3], sa2[3], sa3[3], proj1[3], proj2[3], proj3[3];
  Lin cur1[2], cur2[2], cur3[2], fp32L[3], fp21L[2], fp10L[2], clsL[4];
  { int d[4] = {6, 32, 32, 64};        prep_mlp(stream, st, d_in, 2,  d, 3, sa1); }
  { int d[4] = {67, 96, 96, 128};      prep_mlp(stream, st, d_in, 8,  d, 3, sa2); }
  { int d[4] = {131, 128, 128, 256};   prep_mlp(stream, st, d_in, 14, d, 3, sa3); }
  { int d[4] = {192, 128, 64, 64};     prep_mlp(stream, st, d_in, 65, d, 3, proj1); }
  { int d[4] = {384, 256, 128, 128};   prep_mlp(stream, st, d_in, 71, d, 3, proj2); }
  { int d[4] = {768, 512, 256, 256};   prep_mlp(stream, st, d_in, 77, d, 3, proj3); }
  { int d[3] = {128, 64, 64};          prep_mlp(stream, st, d_in, 83, d, 2, cur1); }
  { int d[3] = {256, 128, 128};        prep_mlp(stream, st, d_in, 87, d, 2, cur2); }
  { int d[3] = {512, 256, 256};        prep_mlp(stream, st, d_in, 91, d, 2, cur3); }
  { int d[4] = {384, 512, 256, 256};   prep_mlp(stream, st, d_in, 95, d, 3, fp32L); }
  { int d[3] = {320, 256, 256};        prep_mlp(stream, st, d_in, 101, d, 2, fp21L); }
  { int d[3] = {256, 256, 128};        prep_mlp(stream, st, d_in, 105, d, 2, fp10L); }
  clsL[0] = prep_lin(stream, st, P(109), nullptr, 128, 128, 128, 0, 0);
  clsL[1] = prep_lin(stream, st, P(110), nullptr, 64, 128, 128, 0, 0);
  clsL[2] = prep_lin(stream, st, P(111), nullptr, 16, 64, 64, 0, 0);
  clsL[3] = prep_lin(stream, st, P(112), nullptr, 3, 16, 16, 0, 0);
  // GAT weights used as f @ W -> transpose into [out,in]
  Lin g1q = prep_lin(stream, st, P(20), nullptr, C1_, C1_, C1_, 1, 0);
  Lin g1k = prep_lin(stream, st, P(21), nullptr, C1_, C1_, C1_, 1, 0);
  Lin g1v = prep_lin(stream, st, P(22), nullptr, C1_, C1_, C1_, 1, 0);
  Lin g2q = prep_lin(stream, st, P(23), nullptr, C2_, C2_, C2_, 1, 0);
  Lin g2k = prep_lin(stream, st, P(24), nullptr, C2_, C2_, C2_, 1, 0);
  Lin g2v = prep_lin(stream, st, P(25), nullptr, C2_, C2_, C2_, 1, 0);
  Lin g3q = prep_lin(stream, st, P(26), nullptr, C3_, C3_, C3_, 1, 0);
  Lin g3k = prep_lin(stream, st, P(27), nullptr, C3_, C3_, C3_, 1, 0);
  Lin g3v = prep_lin(stream, st, P(28), nullptr, C3_, C3_, C3_, 1, 0);
  Mha a1b = prep_mha(stream, st, P(29), P(30), P(31), P(32), C1_);
  Mha a2b = prep_mha(stream, st, P(33), P(34), P(35), P(36), C2_);
  Mha a3b = prep_mha(stream, st, P(37), P(38), P(39), P(40), C3_);
  Mha a1f = prep_mha(stream, st, P(41), P(42), P(43), P(44), C1_);
  Mha a2f = prep_mha(stream, st, P(45), P(46), P(47), P(48), C2_);
  Mha a3f = prep_mha(stream, st, P(49), P(50), P(51), P(52), C3_);
  Mha a1c = prep_mha(stream, st, P(53), P(54), P(55), P(56), C1_);
  Mha a2c = prep_mha(stream, st, P(57), P(58), P(59), P(60), C2_);
  Mha a3c = prep_mha(stream, st, P(61), P(62), P(63), P(64), C3_);

  // ---- per-frame positions ----
  float* posF[T_];
  for (int t = 0; t < T_; ++t) {
    posF[t] = (float*)salloc(st, (size_t)BATCH * NPTS * 3 * sizeof(float));
    size_t tot = (size_t)BATCH * NPTS * 3;
    k_pos_frame<<<(unsigned)((tot + 255) / 256), 256, 0, stream>>>(
        (const float*)d_in[0], posF[t], t, BATCH, NPTS);
  }
  float* pos3 = posF[T_ - 1];

  // ---- hierarchical set abstraction, all frames ----
  float* fea1 = (float*)salloc(st, (size_t)T_ * BATCH * N1_ * C1_ * sizeof(float));
  float* x1   = (float*)salloc(st, (size_t)T_ * BATCH * N1_ * 3 * sizeof(float));
  float* fea2 = (float*)salloc(st, (size_t)T_ * BATCH * N2_ * C2_ * sizeof(float));
  float* x2   = (float*)salloc(st, (size_t)T_ * BATCH * N2_ * 3 * sizeof(float));
  float* fea3 = (float*)salloc(st, (size_t)T_ * BATCH * N3_ * C3_ * sizeof(float));
  float* x3   = (float*)salloc(st, (size_t)T_ * BATCH * N3_ * 3 * sizeof(float));
  for (int t = 0; t < T_; ++t)
    for (int b = 0; b < BATCH; ++b) {
      size_t tb = (size_t)t * BATCH + b;
      const float* p0 = posF[t] + (size_t)b * NPTS * 3;
      run_sa(stream, st, p0, 3, p0, NPTS, N1_, 16, sa1, 3,
             fea1 + tb * N1_ * C1_, x1 + tb * N1_ * 3);
      run_sa(stream, st, fea1 + tb * N1_ * C1_, C1_, x1 + tb * N1_ * 3, N1_,
             N2_, 16, sa2, 3, fea2 + tb * N2_ * C2_, x2 + tb * N2_ * 3);
      run_sa(stream, st, fea2 + tb * N2_ * C2_, C2_, x2 + tb * N2_ * 3, N2_,
             N3_, 8, sa3, 3, fea3 + tb * N3_ * C3_, x3 + tb * N3_ * 3);
    }

  // ---- temporal levels ----
  float* out1 = (float*)salloc(st, (size_t)BATCH * N1_ * C1_ * sizeof(float));
  float* out2 = (float*)salloc(st, (size_t)BATCH * N2_ * C2_ * sizeof(float));
  float* out3 = (float*)salloc(st, (size_t)BATCH * N3_ * C3_ * sizeof(float));
  run_level(stream, st, fea1, x1, N1_, C1_, 32, g1q, g1k, g1v, a1b, a1f, a1c, proj1, cur1, out1);
  run_level(stream, st, fea2, x2, N2_, C2_, 16, g2q, g2k, g2v, a2b, a2f, a2c, proj2, cur2, out2);
  run_level(stream, st, fea3, x3, N3_, C3_, 8,  g3q, g3k, g3v, a3b, a3f, a3c, proj3, cur3, out3);

  // ---- feature propagation (last frame) ----
  const float* x3l = x3 + (size_t)3 * BATCH * N3_ * 3;
  const float* x2l = x2 + (size_t)3 * BATCH * N2_ * 3;
  const float* x1l = x1 + (size_t)3 * BATCH * N1_ * 3;
  float* x2f = (float*)salloc(st, (size_t)BATCH * N2_ * C3_ * sizeof(float));
  run_fp(stream, st, out3, C3_, N3_, x3l, x2l, N2_, out2, C2_, 8, fp32L, 3, x2f, C3_);
  float* x1f = (float*)salloc(st, (size_t)BATCH * N1_ * 256 * sizeof(float));
  run_fp(stream, st, x2f, C3_, N2_, x2l, x1l, N1_, out1, C1_, 16, fp21L, 2, x1f, 256);
  float* x0 = (float*)salloc(st, (size_t)BATCH * NPTS * 128 * sizeof(float));
  run_fp(stream, st, x1f, 256, N1_, x1l, pos3, NPTS, nullptr, 0, 16, fp10L, 2, x0, 128);

  // ---- classifier chain (no relu, no bias) + residual ----
  float* h = run_mlp(stream, st, x0, BATCH * NPTS, 128, clsL, 4, 0);
  int tot = BATCH * NPTS * 3;
  k_add<<<(unsigned)((tot + 255) / 256), 256, 0, stream>>>(pos3, h, (float*)d_out, tot);
}
